// StudentModel_43800076484845
// MI455X (gfx1250) — matrix operand order repaired and checked by reading, not compiled, not run on hardware
//
#include <hip/hip_runtime.h>
#include <hip/hip_bf16.h>
#include <math.h>

#define B_SZ   2048
#define D_SZ   6144
#define H_SZ   2048
#define E_SZ   4
#define NB_SZ  6
#define GH1_SZ 512
#define GH2_SZ 128
#define PH1_SZ 4096
#define PH2_SZ 2048
#define OUT_SZ 2000

typedef __attribute__((ext_vector_type(16))) __bf16 v16bf;
typedef __attribute__((ext_vector_type(8)))  float  v8f;

__device__ __forceinline__ unsigned int pack_bf16x2(float a, float b) {
  unsigned int ua = __float_as_uint(a), ub = __float_as_uint(b);
  ua = (ua + 0x7FFFu + ((ua >> 16) & 1u)) >> 16;   // round-to-nearest-even
  ub = (ub + 0x7FFFu + ((ub >> 16) & 1u)) >> 16;
  return (ua & 0xFFFFu) | (ub << 16);
}

__device__ __forceinline__ float gelu_exact(float x) {
  return 0.5f * x * (1.0f + erff(x * 0.70710678118654752f));
}

// ---------------------------------------------------------------------------
// Tiled bf16 WMMA GEMM:  C[M,N] = epi(A[M,K] @ W[K,N] + bias)
// Block tile 256x128, BK=32. 256 threads = 8 wave32 waves arranged 4x2;
// each wave owns a 64x64 tile = 4x4 grid of v_wmma_f32_16x16x32_bf16
// (16 WMMA per k-step against 8 v16bf fragment loads -> 1 ds_b128 per WMMA).
// LDS holds tiles pre-shuffled into WMMA fragment order (ISA 7.12.2):
//   A 16x32 bf16: lane = (m&15) + 16*((k>>3)&1), elem j = (k&7)|((k>>4)<<3)
//   B 32x16 bf16: lane = (n&15) + 16*((k>>3)&1), elem j = (k&7)|((k>>4)<<3)
// EPI: 0=bias, 1=bias+relu, 2=bias+gelu, 3=bias+residual accumulate into C
// ---------------------------------------------------------------------------
template <int EPI>
__global__ __launch_bounds__(256)
void gemm_bf16_wmma(const float* __restrict__ A, const float* __restrict__ W,
                    const float* __restrict__ bias, float* __restrict__ C,
                    int M, int N, int K) {
  __shared__ __align__(32) unsigned short fragA[2][16][32][16];  // 32 KB
  __shared__ __align__(32) unsigned short fragB[2][8][32][16];   // 16 KB

  const int tid   = threadIdx.x;
  const int ln    = tid & 31;
  const int wv    = tid >> 5;
  const int waveM = wv >> 1;    // 0..3  -> 64 rows each
  const int waveN = wv & 1;     // 0..1  -> 64 cols each
  const int bm    = blockIdx.y;
  const int bn    = blockIdx.x;
  const int nk    = K >> 5;

  v8f zero = {};
  v8f acc[4][4];
#pragma unroll
  for (int i = 0; i < 4; ++i)
#pragma unroll
    for (int j = 0; j < 4; ++j) acc[i][j] = zero;

  // per-thread staging coordinates (fixed across k-steps)
  const int    a_mt   = tid >> 4;          // 0..15 (row tile)
  const int    a_mr   = tid & 15;          // row within tile
  const size_t a_goff = (size_t)(bm * 256 + tid) * (size_t)K;

  const int b_kr   = (tid >> 4) * 2;       // 0..30 (even)
  const int b_n0   = (tid & 15) * 8;       // 0..120
  const int b_gcol = bn * 128 + b_n0;

  auto fill = [&](int buf, int kt) {
    const int k0 = kt << 5;
    // ---- A tile: one full row of 32 fp32 per thread, staged in 2 chunks ----
#pragma unroll
    for (int half = 0; half < 2; ++half) {
      const float4* p = (const float4*)(A + a_goff + (size_t)(k0 + half * 16));
      float4 v0 = p[0], v1 = p[1], v2 = p[2], v3 = p[3];
      float vv[16] = {v0.x, v0.y, v0.z, v0.w, v1.x, v1.y, v1.z, v1.w,
                      v2.x, v2.y, v2.z, v2.w, v3.x, v3.y, v3.z, v3.w};
#pragma unroll
      for (int i = 0; i < 8; ++i) {
        int k    = half * 16 + 2 * i;                 // even
        int lane = a_mr + (((k >> 3) & 1) << 4);
        int j    = (k & 7) | ((k >> 4) << 3);         // even
        *(unsigned int*)&fragA[buf][a_mt][lane][j] =
            pack_bf16x2(vv[2 * i], vv[2 * i + 1]);
      }
    }
    // ---- B tile: 2 k-rows x 8 cols per thread, pack k-pairs ----
    {
      const float* r0 = W + (size_t)(k0 + b_kr) * (size_t)N + b_gcol;
      const float* r1 = r0 + N;
      float4 q00 = make_float4(0.f, 0.f, 0.f, 0.f), q01 = q00;
      float4 q10 = q00, q11 = q00;
      if (b_gcol + 3 < N) { q00 = *(const float4*)r0; q10 = *(const float4*)r1; }
      if (b_gcol + 7 < N) { q01 = *(const float4*)(r0 + 4); q11 = *(const float4*)(r1 + 4); }
      float u0[8] = {q00.x, q00.y, q00.z, q00.w, q01.x, q01.y, q01.z, q01.w};
      float u1[8] = {q10.x, q10.y, q10.z, q10.w, q11.x, q11.y, q11.z, q11.w};
      const int lhalf = ((b_kr >> 3) & 1) << 4;
      const int j     = (b_kr & 7) | ((b_kr >> 4) << 3); // even
#pragma unroll
      for (int i = 0; i < 8; ++i) {
        int n    = b_n0 + i;
        int lane = (n & 15) + lhalf;
        int nt   = n >> 4;
        *(unsigned int*)&fragB[buf][nt][lane][j] = pack_bf16x2(u0[i], u1[i]);
      }
    }
  };

  fill(0, 0);
  __syncthreads();

  for (int kt = 0; kt < nk; ++kt) {
    const int cur = kt & 1;
    if (kt + 1 < nk) {
      fill(cur ^ 1, kt + 1);                  // stage next while computing cur
      if (kt + 2 < nk) {                      // pull k-distance-2 toward L2/WGP$
        const int kp = (kt + 2) << 5;
        __builtin_prefetch(A + a_goff + kp, 0, 0);
        __builtin_prefetch(W + (size_t)(kp + b_kr) * (size_t)N + b_gcol, 0, 0);
      }
    }

    v16bf af[4], bf4[4];
#pragma unroll
    for (int mf = 0; mf < 4; ++mf)
      af[mf] = *(const v16bf*)&fragA[cur][waveM * 4 + mf][ln][0];
#pragma unroll
    for (int nf = 0; nf < 4; ++nf)
      bf4[nf] = *(const v16bf*)&fragB[cur][waveN * 4 + nf][ln][0];

#pragma unroll
    for (int mf = 0; mf < 4; ++mf)
#pragma unroll
      for (int nf = 0; nf < 4; ++nf)
        acc[mf][nf] = __builtin_amdgcn_wmma_f32_16x16x32_bf16(
            false, af[mf], false, bf4[nf], (short)0, acc[mf][nf], false, false);

    __syncthreads();
  }

  // ---- epilogue: C/D layout — VGPR r = row (mf*16 + r + 8*(ln>>4)), col = ln&15
#pragma unroll
  for (int mf = 0; mf < 4; ++mf) {
#pragma unroll
    for (int nf = 0; nf < 4; ++nf) {
      int col = bn * 128 + waveN * 64 + nf * 16 + (ln & 15);
      if (col < N) {
        float bv   = bias ? bias[col] : 0.0f;
        int   row0 = bm * 256 + waveM * 64 + mf * 16 + ((ln >> 4) << 3);
        float* cp  = C + (size_t)row0 * (size_t)N + col;
#pragma unroll
        for (int r = 0; r < 8; ++r) {
          float v = acc[mf][nf][r] + bv;
          if (EPI == 1) v = v > 0.0f ? v : 0.0f;
          if (EPI == 2) v = gelu_exact(v);
          if (EPI == 3) v += cp[(size_t)r * N];
          cp[(size_t)r * N] = v;
        }
      }
    }
  }
}

// ---------------------------------------------------------------------------
// Replicate x into 4 per-expert state buffers (read once, write 4x).
// ---------------------------------------------------------------------------
__global__ void replicate4_kernel(const float* __restrict__ x,
                                  float* __restrict__ h, int n4) {
  int i = blockIdx.x * blockDim.x + threadIdx.x;
  if (i < n4) {
    float4 v = ((const float4*)x)[i];
    ((float4*)h)[i]          = v;
    ((float4*)h)[i + n4]     = v;
    ((float4*)h)[i + 2 * n4] = v;
    ((float4*)h)[i + 3 * n4] = v;
  }
}

// ---------------------------------------------------------------------------
// Row LayerNorm (D=6144), one block per row, 256-thread tree reduction.
// ---------------------------------------------------------------------------
__global__ __launch_bounds__(256)
void layernorm_kernel(const float* __restrict__ x, const float* __restrict__ g,
                      const float* __restrict__ b, float* __restrict__ y) {
  const int row = blockIdx.x;
  const int tid = threadIdx.x;
  const float* xr = x + (size_t)row * D_SZ;
  float s = 0.f, s2 = 0.f;
  for (int i = tid; i < D_SZ; i += 256) { float v = xr[i]; s += v; s2 += v * v; }
  __shared__ float sh[256], sh2[256];
  sh[tid] = s; sh2[tid] = s2;
  __syncthreads();
  for (int off = 128; off > 0; off >>= 1) {
    if (tid < off) { sh[tid] += sh[tid + off]; sh2[tid] += sh2[tid + off]; }
    __syncthreads();
  }
  float mean = sh[0] * (1.0f / D_SZ);
  float var  = sh2[0] * (1.0f / D_SZ) - mean * mean;
  float rs   = rsqrtf(var + 1e-5f);
  float* yr  = y + (size_t)row * D_SZ;
  for (int i = tid; i < D_SZ; i += 256)
    yr[i] = (xr[i] - mean) * rs * g[i] + b[i];
}

// ---------------------------------------------------------------------------
// Gate logits: [B,128] @ [128,4] + b3 + expert_bias (tiny; one thread per b,e)
// ---------------------------------------------------------------------------
__global__ void gate_logits_kernel(const float* __restrict__ g2,
                                   const float* __restrict__ w3,
                                   const float* __restrict__ b3,
                                   const float* __restrict__ ebias,
                                   float* __restrict__ logits) {
  int t = blockIdx.x * blockDim.x + threadIdx.x;
  if (t >= B_SZ * E_SZ) return;
  int bi = t >> 2, e = t & 3;
  const float* gr = g2 + (size_t)bi * GH2_SZ;
  float s = 0.f;
  for (int k = 0; k < GH2_SZ; ++k) s += gr[k] * w3[k * E_SZ + e];
  logits[t] = s + b3[e] + ebias[e];
}

// ---------------------------------------------------------------------------
// Softmax over 4 + top-2 (JAX tie-break: lower index first) + renormalize.
// ---------------------------------------------------------------------------
__global__ void topk_kernel(const float* __restrict__ logits,
                            float* __restrict__ out_w, int* __restrict__ out_i,
                            float* __restrict__ wtop, int* __restrict__ itop) {
  int bi = blockIdx.x * blockDim.x + threadIdx.x;
  if (bi >= B_SZ) return;
  float l[4], w[4], mx = -1e30f;
  for (int e = 0; e < 4; ++e) { l[e] = logits[bi * 4 + e]; mx = fmaxf(mx, l[e]); }
  float sum = 0.f;
  for (int e = 0; e < 4; ++e) { w[e] = expf(l[e] - mx); sum += w[e]; }
  float inv = 1.0f / sum;
  for (int e = 0; e < 4; ++e) { w[e] *= inv; out_w[bi * 4 + e] = w[e]; }
  int i0 = 0;
  for (int e = 1; e < 4; ++e) if (w[e] > w[i0]) i0 = e;
  int i1 = -1;
  for (int e = 0; e < 4; ++e) if (e != i0 && (i1 < 0 || w[e] > w[i1])) i1 = e;
  float ti = 1.0f / (w[i0] + w[i1]);
  wtop[bi * 2] = w[i0] * ti; wtop[bi * 2 + 1] = w[i1] * ti;
  itop[bi * 2] = i0;         itop[bi * 2 + 1] = i1;
  out_i[bi * 2] = i0;        out_i[bi * 2 + 1] = i1;
}

// ---------------------------------------------------------------------------
// Top-2 weighted combine of expert outputs.
// ---------------------------------------------------------------------------
__global__ void combine_kernel(const float* __restrict__ h,
                               const float* __restrict__ wtop,
                               const int* __restrict__ itop,
                               float* __restrict__ comb) {
  int bi = blockIdx.y;
  int d  = blockIdx.x * 256 + threadIdx.x;
  float w0 = wtop[bi * 2], w1 = wtop[bi * 2 + 1];
  int   i0 = itop[bi * 2], i1 = itop[bi * 2 + 1];
  size_t bd = (size_t)B_SZ * D_SZ;
  size_t o  = (size_t)bi * D_SZ + d;
  comb[o] = w0 * h[(size_t)i0 * bd + o] + w1 * h[(size_t)i1 * bd + o];
}

// ---------------------------------------------------------------------------
extern "C" void kernel_launch(void* const* d_in, const int* in_sizes, int n_in,
                              void* d_out, int out_size, void* d_ws, size_t ws_size,
                              hipStream_t stream) {
  (void)in_sizes; (void)n_in; (void)out_size; (void)ws_size;
  const float* x     = (const float*)d_in[0];
  const float* gw1   = (const float*)d_in[1];
  const float* gb1   = (const float*)d_in[2];
  const float* gw2   = (const float*)d_in[3];
  const float* gb2   = (const float*)d_in[4];
  const float* gw3   = (const float*)d_in[5];
  const float* gb3   = (const float*)d_in[6];
  const float* ebias = (const float*)d_in[7];
  const float* lns   = (const float*)d_in[8];
  const float* lnb   = (const float*)d_in[9];
  const float* win   = (const float*)d_in[10];
  const float* bin   = (const float*)d_in[11];
  const float* wout  = (const float*)d_in[12];
  const float* bout  = (const float*)d_in[13];
  const float* hw1   = (const float*)d_in[14];
  const float* hb1   = (const float*)d_in[15];
  const float* hw2   = (const float*)d_in[16];
  const float* hb2   = (const float*)d_in[17];
  const float* hw3   = (const float*)d_in[18];
  const float* hb3   = (const float*)d_in[19];

  float* ws = (float*)d_ws;
  size_t off = 0;
  float* h      = ws + off; off += (size_t)E_SZ * B_SZ * D_SZ;
  float* t0     = ws + off; off += (size_t)B_SZ * D_SZ;
  float* t1     = ws + off; off += (size_t)B_SZ * H_SZ;
  float* g1     = ws + off; off += (size_t)B_SZ * GH1_SZ;
  float* g2b    = ws + off; off += (size_t)B_SZ * GH2_SZ;
  float* logits = ws + off; off += (size_t)B_SZ * E_SZ;
  float* wtop   = ws + off; off += (size_t)B_SZ * 2;
  int*   itop   = (int*)(ws + off); off += (size_t)B_SZ * 2;
  float* comb   = ws + off; off += (size_t)B_SZ * D_SZ;
  float* h1     = ws + off; off += (size_t)B_SZ * PH1_SZ;
  float* h2     = ws + off; off += (size_t)B_SZ * PH2_SZ;

  float* spec  = (float*)d_out;                         // [B, 2000]
  float* out_w = spec + (size_t)B_SZ * OUT_SZ;          // [B, 4]
  int*   out_i = (int*)(out_w + (size_t)B_SZ * E_SZ);   // [B, 2] int32

  dim3 blk(256);
  const int gy = B_SZ / 256;                            // 256-row block tiles

  // per-expert residual state = x
  int n4 = B_SZ * D_SZ / 4;
  replicate4_kernel<<<dim3((n4 + 255) / 256), blk, 0, stream>>>(x, h, n4);

  // gate MLP
  gemm_bf16_wmma<1><<<dim3(GH1_SZ / 128, gy), blk, 0, stream>>>(x, gw1, gb1, g1, B_SZ, GH1_SZ, D_SZ);
  gemm_bf16_wmma<1><<<dim3(GH2_SZ / 128, gy), blk, 0, stream>>>(g1, gw2, gb2, g2b, B_SZ, GH2_SZ, GH1_SZ);
  gate_logits_kernel<<<dim3((B_SZ * E_SZ + 255) / 256), blk, 0, stream>>>(g2b, gw3, gb3, ebias, logits);
  topk_kernel<<<dim3((B_SZ + 255) / 256), blk, 0, stream>>>(logits, out_w, out_i, wtop, itop);

  // dense experts: 24 residual blocks, 48 big WMMA GEMMs
  for (int e = 0; e < E_SZ; ++e) {
    float* he = h + (size_t)e * B_SZ * D_SZ;
    for (int nb = 0; nb < NB_SZ; ++nb) {
      size_t pi = (size_t)(e * NB_SZ + nb);
      layernorm_kernel<<<dim3(B_SZ), blk, 0, stream>>>(he, lns + pi * D_SZ, lnb + pi * D_SZ, t0);
      gemm_bf16_wmma<2><<<dim3(H_SZ / 128, gy), blk, 0, stream>>>(
          t0, win + pi * (size_t)D_SZ * H_SZ, bin + pi * H_SZ, t1, B_SZ, H_SZ, D_SZ);
      gemm_bf16_wmma<3><<<dim3(D_SZ / 128, gy), blk, 0, stream>>>(
          t1, wout + pi * (size_t)H_SZ * D_SZ, bout + pi * D_SZ, he, B_SZ, D_SZ, H_SZ);
    }
  }

  // top-2 combine
  combine_kernel<<<dim3(D_SZ / 256, B_SZ), blk, 0, stream>>>(h, wtop, itop, comb);

  // prediction head
  gemm_bf16_wmma<2><<<dim3(PH1_SZ / 128, gy), blk, 0, stream>>>(comb, hw1, hb1, h1, B_SZ, PH1_SZ, D_SZ);
  gemm_bf16_wmma<2><<<dim3(PH2_SZ / 128, gy), blk, 0, stream>>>(h1, hw2, hb2, h2, B_SZ, PH2_SZ, PH1_SZ);
  gemm_bf16_wmma<0><<<dim3((OUT_SZ + 127) / 128, gy), blk, 0, stream>>>(h2, hw3, hb3, spec, B_SZ, OUT_SZ, PH2_SZ);
}
